// TTLinear_45947560132801
// MI455X (gfx1250) — matrix operand order repaired and checked by reading, not compiled, not run on hardware
//
#include <hip/hip_runtime.h>
#include <hip/hip_bf16.h>

// Fused TT-linear for MI455X (gfx1250, wave32, WMMA), round 4.
// bf16 operands, f32 accumulation. W0 folded into W1 per-o0 (WMMA precompute).
// Convert+store of each WMMA D-tile is software-pipelined one iteration behind
// its producer to cover the WMMA->VALU RAW hazard (removes v_nop stalls).
// 2 barriers per inner iteration; conflict-free padded LDS rows; 318 KB LDS.

typedef __attribute__((ext_vector_type(16))) __bf16 v16bf;
typedef __attribute__((ext_vector_type(8)))  __bf16 v8bf;
typedef __attribute__((ext_vector_type(8)))  float  v8f;

#define NB   16
#define RSTR 136   // 128-K row stride (272B = 68 dwords -> bank stride 4n)
#define XSTR 72    // 64-K row stride (144B = 36 dwords, 16B aligned)
#define WSTR 40    // small-weight row stride (80B = 20 dwords)

#define OFF_W0TT 0                          // [o0*16+i0(pad16)][r1(40 pad)]      5120
#define OFF_W1L  (OFF_W0TT + 8*16*WSTR)     // [m][k'=i1*16+r1 (136 pad)]        17408
#define OFF_W2L  (OFF_W1L + 128*RSTR)       // [m][k'=i2*16+r2 (136 pad)]        17408
#define OFF_W3T  (OFF_W2L + 128*RSTR)       // [i3*16+o3][r3 (40 pad)]            5120
#define OFF_XS   (OFF_W3T + 8*16*WSTR)      // [((i2*8+i3)*16+b)][i1*8+i0 (72)]  73728
#define OFF_W01  (OFF_XS  + 64*16*XSTR)     // [(o1*16+r2)][i1*8+i0 (72 pad)]     9216
#define OFF_BA   (OFF_W01 + 128*XSTR)       // T3: [o1*16+b][o2*16+r3 (136)]     17408
#define OFF_BB   (OFF_BA  + 8*16*RSTR)      // T2: [o1*16+b][i2*16+r2 (136)]     17408
#define OFF_ZP   (OFF_BB  + 8*16*RSTR)      // 32 zeros
#define N_ELEMS  (OFF_ZP + 32)              // 162848 elems
#define SMEM_BYTES (N_ELEMS * 2)            // 325696 B = 318 KB (<= 320 KB WGP LDS)

__device__ __forceinline__ v8f wmma_bf16(v16bf a, v16bf b, v8f c) {
  return __builtin_amdgcn_wmma_f32_16x16x32_bf16(false, a, false, b, (short)0, c,
                                                 false, false);
}

__device__ __forceinline__ v16bf cat8(v8bf lo, v8bf hi) {
  union { v16bf v; v8bf h[2]; } u;
  u.h[0] = lo; u.h[1] = hi;
  return u.v;
}

__device__ __forceinline__ v8bf ldsv8(const __bf16* p) { return *(const v8bf*)p; }

__device__ __forceinline__ v8bf cvt8(v8f d) {
  v8bf r;
  #pragma unroll
  for (int e = 0; e < 8; ++e) r[e] = (__bf16)d[e];
  return r;
}

__global__ __launch_bounds__(256)
void TTLinear_fused_kernel(const float* __restrict__ x,
                           const float* __restrict__ w0,
                           const float* __restrict__ w1,
                           const float* __restrict__ w2,
                           const float* __restrict__ w3,
                           const float* __restrict__ bias,
                           float* __restrict__ out)
{
  extern __shared__ __align__(32) char smem[];
  __bf16* S    = (__bf16*)smem;
  __bf16* W0TT = S + OFF_W0TT;
  __bf16* W1L  = S + OFF_W1L;
  __bf16* W2L  = S + OFF_W2L;
  __bf16* W3T  = S + OFF_W3T;
  __bf16* Xs   = S + OFF_XS;
  __bf16* W01  = S + OFF_W01;   // per-o0 folded W0*W1, rebuilt each o0 iteration
  __bf16* bufA = S + OFF_BA;
  __bf16* bufB = S + OFF_BB;
  __bf16* ZP   = S + OFF_ZP;

  const int tid  = threadIdx.x;
  const int lane = tid & 31;
  const int wv   = tid >> 5;     // wave 0..7 == its M-tile / o1-group
  const int n    = lane & 15;    // WMMA B/D column
  const int h    = lane >> 4;    // lane half (A and B: K-run offset 8h; S3: K=16h..)
  const int m16  = lane & 15;    // WMMA A row within M-tile
  const long bbase = (long)blockIdx.x * NB;

  const __bf16 zbf = (__bf16)0.0f;
  const v8f zero8f = { 0.f,0.f,0.f,0.f,0.f,0.f,0.f,0.f };

  // ---------------- phase 1: big weights (K-permuted), zero pads ----------------
  for (int i = tid; i < 16384; i += 256) {            // w1[m][r1*8+i1] -> [m][i1*16+r1]
    int m = i >> 7, col = i & 127;
    W1L[m * RSTR + ((col & 7) << 4) + (col >> 3)] = (__bf16)w1[i];
  }
  for (int i = tid; i < 16384; i += 256) {            // w2[m][r2*8+i2] -> [m][i2*16+r2]
    int m = i >> 7, col = i & 127;
    W2L[m * RSTR + ((col & 7) << 4) + (col >> 3)] = (__bf16)w2[i];
  }
  for (int i = tid; i < 8*16*WSTR; i += 256) { W0TT[i] = zbf; W3T[i] = zbf; }
  if (tid < 32) ZP[tid] = zbf;
  __syncthreads();

  // ---------------- phase 2: small weights (transposed) + X tile ----------------
  for (int i = tid; i < 1024; i += 256) {             // w0[(o0*16+r1)][i0] -> [o0,i0][r1]
    int row = i >> 3, i0 = i & 7;
    int o0 = row >> 4, r1 = row & 15;
    W0TT[(((o0 << 4) + i0)) * WSTR + r1] = (__bf16)w0[i];
  }
  for (int i = tid; i < 1024; i += 256) {             // w3[o3][r3*8+i3] -> [i3,o3][r3]
    int o3 = i >> 7, col = i & 127;
    int r3 = col >> 3, i3c = col & 7;
    W3T[((i3c << 4) + o3) * WSTR + r3] = (__bf16)w3[i];
  }
  const float4* xv = (const float4*)(x + bbase * 4096);
  for (int q = tid; q < 16384; q += 256) {            // 16 samples * 4096 / 4
    float4 vx = xv[q];
    int fi = q << 2;
    int b  = fi >> 12;
    int f  = fi & 4095;
    int i0 = f >> 9, i1 = (f >> 6) & 7, i2 = (f >> 3) & 7, i3b = f & 7;
    int kcol  = (i1 << 3) + i0;
    int rbase = (i2 << 3) + i3b;
    Xs[((rbase + 0) * 16 + b) * XSTR + kcol] = (__bf16)vx.x;
    Xs[((rbase + 1) * 16 + b) * XSTR + kcol] = (__bf16)vx.y;
    Xs[((rbase + 2) * 16 + b) * XSTR + kcol] = (__bf16)vx.z;
    Xs[((rbase + 3) * 16 + b) * XSTR + kcol] = (__bf16)vx.w;
  }
  __syncthreads();

  // ---------------- preload per-wave invariant W2 A-operands (M-tile = wv) -------
  v16bf a2[4];
  {
    const __bf16* p = W2L + ((wv << 4) + m16) * RSTR + (h << 3);
    #pragma unroll
    for (int kt = 0; kt < 4; ++kt)
      a2[kt] = cat8(ldsv8(p + (kt << 5)), ldsv8(p + (kt << 5) + 16));
  }

  // ---------------- main contraction ----------------
  for (int o0 = 0; o0 < 8; ++o0) {
    // ---- fold W0[o0] into W1 with WMMA: W01[(o1,r2)][i1*8+i0] (wave owns rows wv*16..)
    // store pipelined one iteration behind its producing WMMA
    {
      const __bf16* pb0 = W0TT + ((o0 << 4) + n) * WSTR + (h << 3);
      v16bf b0 = cat8(ldsv8(pb0), ldsv8(pb0 + 16));
      v8f dp;
      #pragma unroll
      for (int j = 0; j < 8; ++j) {   // j = i1
        v16bf aP = cat8(ldsv8(W1L + ((wv << 4) + m16) * RSTR + (j << 4) + (h << 3)),
                        ldsv8(ZP));
        v8f d = wmma_bf16(aP, b0, zero8f);
        if (j > 0 && n < 8) {
          #pragma unroll
          for (int e = 0; e < 8; ++e)
            W01[((wv << 4) + (h << 3) + e) * XSTR + ((j - 1) << 3) + n] = (__bf16)dp[e];
        }
        dp = d;
      }
      if (n < 8) {
        #pragma unroll
        for (int e = 0; e < 8; ++e)
          W01[((wv << 4) + (h << 3) + e) * XSTR + (7 << 3) + n] = (__bf16)dp[e];
      }
    }
    // wave reads only its own rows -> in-order LDS per wave, no barrier needed
    v16bf a01[2];
    {
      const __bf16* p = W01 + ((wv << 4) + m16) * XSTR + (h << 3);
      a01[0] = cat8(ldsv8(p), ldsv8(p + 16));
      a01[1] = cat8(ldsv8(p + 32), ldsv8(p + 48));
    }

    v8f acc[8];
    #pragma unroll
    for (int j = 0; j < 8; ++j) acc[j] = zero8f;

    for (int i3 = 0; i3 < 8; ++i3) {
      v16bf a3;   // W3 slice at i3 (K = r3, zero-padded in LDS)
      {
        const __bf16* p = W3T + ((i3 << 4) + m16) * WSTR + (h << 4);
        a3 = cat8(ldsv8(p), ldsv8(p + 8));
      }

      // ---- S1: T2[(o1=wv,r2), i2=j, b] = W01 @ X   (K = 64, no padding)
      {
        const int sb = ((wv << 4) + n) * RSTR + (h << 3);
        v8f cp;
        #pragma unroll
        for (int j = 0; j < 8; ++j) {
          const __bf16* pb = Xs + ((((j << 3) + i3) << 4) + n) * XSTR + (h << 3);
          v8f c = wmma_bf16(a01[0], cat8(ldsv8(pb),      ldsv8(pb + 16)), zero8f);
          c     = wmma_bf16(a01[1], cat8(ldsv8(pb + 32), ldsv8(pb + 48)), c);
          if (j > 0) *(v8bf*)(bufB + sb + ((j - 1) << 4)) = cvt8(cp);
          cp = c;
        }
        *(v8bf*)(bufB + sb + (7 << 4)) = cvt8(cp);
      }
      __syncthreads();

      // ---- S2: T3[(o2=wv,r3), o1=j, b] = W2 @ T2   (K = 128)
      {
        const int sb = (wv << 4) + (h << 3);
        v8f cp;
        #pragma unroll
        for (int j = 0; j < 8; ++j) {
          const __bf16* pb = bufB + ((j << 4) + n) * RSTR + (h << 3);
          v8f c = zero8f;
          #pragma unroll
          for (int kt = 0; kt < 4; ++kt)
            c = wmma_bf16(a2[kt], cat8(ldsv8(pb + (kt << 5)), ldsv8(pb + (kt << 5) + 16)), c);
          if (j > 0) *(v8bf*)(bufA + (((j - 1) << 4) + n) * RSTR + sb) = cvt8(cp);
          cp = c;
        }
        *(v8bf*)(bufA + ((7 << 4) + n) * RSTR + sb) = cvt8(cp);
      }
      __syncthreads();

      // ---- S3: acc[(o1=wv, o2=j)] += W3[:,r3,i3] @ T3   (K=16 pad 32)
      // (no trailing barrier: next S1 only writes bufB; the post-S1 barrier
      //  orders this S3 against the next S2's bufA writes)
      #pragma unroll
      for (int j = 0; j < 8; ++j) {
        const __bf16* pb3 = h ? ZP : (bufA + ((wv << 4) + n) * RSTR + (j << 4));
        acc[j] = wmma_bf16(a3, cat8(ldsv8(pb3), ldsv8(pb3 + 8)), acc[j]);
      }
    } // i3

    // ---- write out (+bias): valid rows o3 = 0..7 live in lane-half h==0
    if (h == 0) {
      #pragma unroll
      for (int j = 0; j < 8; ++j) {
        int fo = (o0 << 9) + (wv << 6) + (j << 3);
        float* po = out + (bbase + n) * 4096 + fo;
        const float4* pb4 = (const float4*)(bias + fo);
        float4 b0 = pb4[0], b1 = pb4[1];
        float4 q0, q1;
        q0.x = acc[j][0] + b0.x; q0.y = acc[j][1] + b0.y;
        q0.z = acc[j][2] + b0.z; q0.w = acc[j][3] + b0.w;
        q1.x = acc[j][4] + b1.x; q1.y = acc[j][5] + b1.y;
        q1.z = acc[j][6] + b1.z; q1.w = acc[j][7] + b1.w;
        *(float4*)po = q0;
        *(float4*)(po + 4) = q1;
      }
    }
  } // o0
}

extern "C" void kernel_launch(void* const* d_in, const int* in_sizes, int n_in,
                              void* d_out, int out_size, void* d_ws, size_t ws_size,
                              hipStream_t stream) {
  (void)in_sizes; (void)n_in; (void)out_size; (void)d_ws; (void)ws_size;
  const float* x    = (const float*)d_in[0];
  const float* w0   = (const float*)d_in[1];
  const float* w1   = (const float*)d_in[2];
  const float* w2   = (const float*)d_in[3];
  const float* w3   = (const float*)d_in[4];
  const float* bias = (const float*)d_in[5];
  float* out = (float*)d_out;

  hipFuncSetAttribute((const void*)TTLinear_fused_kernel,
                      hipFuncAttributeMaxDynamicSharedMemorySize, SMEM_BYTES);
  dim3 grid(16384 / NB);   // 1024 tiles of 16 samples
  dim3 block(256);         // 8 wave32 waves
  TTLinear_fused_kernel<<<grid, block, SMEM_BYTES, stream>>>(x, w0, w1, w2, w3, bias, out);
}
